// EEGLstmNet_42N_200_59760174956970
// MI455X (gfx1250) — matrix operand order repaired and checked by reading, not compiled, not run on hardware
//
#include <hip/hip_runtime.h>
#include <hip/hip_bf16.h>
#include <cstdint>

typedef _Float16 v16h __attribute__((ext_vector_type(16)));
typedef _Float16 v8h  __attribute__((ext_vector_type(8)));
typedef float    v8f  __attribute__((ext_vector_type(8)));

#define NB 2048          // batch
#define NT 200           // time
#define NCH 64           // channels
#define NH 120           // LSTM hidden
#define HSTR 136         // LDS h stride (halves), 120->128 pad + 8
#define BSTR 144         // LDS Whh^T stride (halves), 128 + 16 pad

__device__ __forceinline__ float eluf(float x) { return x > 0.f ? x : (expf(x) - 1.f); }
__device__ __forceinline__ float sigf(float x) { return 1.f / (1.f + expf(-x)); }

// ---------------- workspace layout (float offsets) ----------------
#define OFF_S1    0        // 8:  conv1 stats (sum4, sumsq4)
#define OFF_SC1   8
#define OFF_SH1   12
#define OFF_S2    16       // 16: conv2 stats
#define OFF_SC2   32
#define OFF_SH2   40
#define OFF_S3    48       // 16: convp stats
#define OFF_SC3   64
#define OFF_SH3   72
#define OFF_CACC  80       // 8:  folded head constants
#define OFF_SO    88       // 16: pre-output stats
#define OFF_SCO   104
#define OFF_SHO   112
#define OFF_A     128      // 1920: folded head weights
#define OFF_TMP2  2048               // 2048*8*200
#define OFF_TMP4  (OFF_TMP2 + 2048*8*200)   // 2048*8*50
#define OFF_SEQ   (OFF_TMP4 + 2048*8*50)    // 8*10*2048  [branch][t][batch]
#define OFF_FEATS (OFF_SEQ + 8*10*2048)     // 8*240*2048 [branch][feat][batch]
#define OFF_ZB    (OFF_FEATS + 8*240*2048)  // 8*2048

// ================= K1: conv1 + elu, BN1 stats =================
__global__ void __launch_bounds__(256)
k_conv1_stats(const float* __restrict__ x, const float* __restrict__ w1,
              const float* __restrict__ b1, float* __restrict__ stats1) {
  extern __shared__ float smA[];
  float* xs   = smA;          // 200*64
  float* w1s  = xs + 12800;   // 404
  float* sacc = w1s + 404;    // 8
  const int b = blockIdx.x, tid = threadIdx.x;
  for (int i = tid; i < 12800; i += 256) xs[i] = x[(size_t)b * 12800 + i];
  for (int i = tid; i < 404; i += 256) w1s[i] = w1[i];
  if (tid < 8) sacc[tid] = 0.f;
  __syncthreads();
  const float cb0 = b1[0], cb1 = b1[1], cb2 = b1[2], cb3 = b1[3];
  float s[4] = {0, 0, 0, 0}, ss[4] = {0, 0, 0, 0};
  for (int p = tid; p < 12800; p += 256) {
    const int t = p >> 6, ch = p & 63;
    float a0 = cb0, a1 = cb1, a2 = cb2, a3 = cb3;
    for (int kk = 0; kk < 101; ++kk) {
      const int tt = t + kk - 50;
      if (tt >= 0 && tt < NT) {
        const float v = xs[tt * 64 + ch];
        a0 += v * w1s[kk];       a1 += v * w1s[101 + kk];
        a2 += v * w1s[202 + kk]; a3 += v * w1s[303 + kk];
      }
    }
    float e;
    e = eluf(a0); s[0] += e; ss[0] += e * e;
    e = eluf(a1); s[1] += e; ss[1] += e * e;
    e = eluf(a2); s[2] += e; ss[2] += e * e;
    e = eluf(a3); s[3] += e; ss[3] += e * e;
  }
  for (int c = 0; c < 4; ++c) { atomicAdd(&sacc[c], s[c]); atomicAdd(&sacc[4 + c], ss[c]); }
  __syncthreads();
  if (tid < 8) atomicAdd(&stats1[tid], sacc[tid]);
}

// ============ generic BN finalize: sc = g*rsqrt(v), sh = b - m*sc ============
__global__ void k_finalize(const float* __restrict__ stats, const float* __restrict__ g,
                           const float* __restrict__ beta, float* __restrict__ sc,
                           float* __restrict__ sh, int n, float cnt) {
  const int c = threadIdx.x;
  if (c < n) {
    const float m = stats[c] / cnt;
    const float v = stats[n + c] / cnt - m * m;
    const float s = g[c] * rsqrtf(v);
    sc[c] = s;
    sh[c] = beta[c] - m * s;
  }
}

// ===== K3: recompute conv1+elu, apply BN1, conv2(groups=4)+elu, BN2 stats =====
__global__ void __launch_bounds__(256)
k_conv12(const float* __restrict__ x, const float* __restrict__ w1, const float* __restrict__ b1,
         const float* __restrict__ sc1, const float* __restrict__ sh1,
         const float* __restrict__ w2, const float* __restrict__ b2,
         float* __restrict__ tmp2, float* __restrict__ stats2) {
  extern __shared__ float smB[];
  float* xs   = smB;          // 12800
  float* w1s  = xs + 12800;   // 404
  float* w2s  = w1s + 404;    // 8*64
  float* o2   = w2s + 512;    // 8*200
  float* sacc = o2 + 1600;    // 16
  const int b = blockIdx.x, tid = threadIdx.x;
  for (int i = tid; i < 12800; i += 256) xs[i] = x[(size_t)b * 12800 + i];
  for (int i = tid; i < 404; i += 256) w1s[i] = w1[i];
  for (int i = tid; i < 512; i += 256) w2s[i] = w2[i];
  for (int i = tid; i < 1600; i += 256) o2[i] = 0.f;
  if (tid < 16) sacc[tid] = 0.f;
  __syncthreads();
  const float cb0 = b1[0], cb1 = b1[1], cb2 = b1[2], cb3 = b1[3];
  const float s0 = sc1[0], s1 = sc1[1], s2 = sc1[2], s3 = sc1[3];
  const float h0 = sh1[0], h1 = sh1[1], h2 = sh1[2], h3 = sh1[3];
  for (int p = tid; p < 12800; p += 256) {
    const int t = p >> 6, ch = p & 63;
    float a0 = cb0, a1 = cb1, a2 = cb2, a3 = cb3;
    for (int kk = 0; kk < 101; ++kk) {
      const int tt = t + kk - 50;
      if (tt >= 0 && tt < NT) {
        const float v = xs[tt * 64 + ch];
        a0 += v * w1s[kk];       a1 += v * w1s[101 + kk];
        a2 += v * w1s[202 + kk]; a3 += v * w1s[303 + kk];
      }
    }
    float en[4];
    en[0] = eluf(a0) * s0 + h0; en[1] = eluf(a1) * s1 + h1;
    en[2] = eluf(a2) * s2 + h2; en[3] = eluf(a3) * s3 + h3;
    #pragma unroll
    for (int c2 = 0; c2 < 8; ++c2)
      atomicAdd(&o2[c2 * 200 + t], en[c2 >> 1] * w2s[c2 * 64 + ch]);
  }
  __syncthreads();
  for (int t = tid; t < 200; t += 256) {
    for (int c2 = 0; c2 < 8; ++c2) {
      const float e2 = eluf(o2[c2 * 200 + t] + b2[c2]);
      tmp2[((size_t)b * 8 + c2) * 200 + t] = e2;
      atomicAdd(&sacc[c2], e2);
      atomicAdd(&sacc[8 + c2], e2 * e2);
    }
  }
  __syncthreads();
  if (tid < 16) atomicAdd(&stats2[tid], sacc[tid]);
}

// ===== K5: BN2 apply, maxpool4, conv3(depthwise)+elu, convp(8x8)+elu, BN3 stats =====
__global__ void __launch_bounds__(256)
k_stage2(const float* __restrict__ tmp2, const float* __restrict__ sc2, const float* __restrict__ sh2,
         const float* __restrict__ w3, const float* __restrict__ b3,
         const float* __restrict__ wp, const float* __restrict__ bp,
         float* __restrict__ tmp4, float* __restrict__ stats3) {
  extern __shared__ float smC[];
  float* e2s  = smC;          // 8*200
  float* p1   = e2s + 1600;   // 8*50
  float* c3   = p1 + 400;     // 8*50
  float* w3s  = c3 + 400;     // 8*25
  float* wps  = w3s + 200;    // 8*8
  float* sacc = wps + 64;     // 16
  const int b = blockIdx.x, tid = threadIdx.x;
  for (int i = tid; i < 200; i += 256) w3s[i] = w3[i];
  for (int i = tid; i < 64; i += 256) wps[i] = wp[i];
  if (tid < 16) sacc[tid] = 0.f;
  for (int i = tid; i < 1600; i += 256) {
    const int c = i / 200;
    e2s[i] = tmp2[(size_t)b * 1600 + i] * sc2[c] + sh2[c];
  }
  __syncthreads();
  for (int i = tid; i < 400; i += 256) {       // maxpool k=4
    const int c = i / 50, t = i - c * 50;
    const float* row = e2s + c * 200 + t * 4;
    float m = row[0];
    for (int q = 1; q < 4; ++q) m = row[q] > m ? row[q] : m;
    p1[i] = m;
  }
  __syncthreads();
  for (int i = tid; i < 400; i += 256) {       // depthwise 25-tap + elu
    const int c = i / 50, t = i - c * 50;
    float acc = b3[c];
    for (int kk = 0; kk < 25; ++kk) {
      const int tt = t + kk - 12;
      if (tt >= 0 && tt < 50) acc += p1[c * 50 + tt] * w3s[c * 25 + kk];
    }
    c3[i] = eluf(acc);
  }
  __syncthreads();
  for (int i = tid; i < 400; i += 256) {       // pointwise 8x8 + elu
    const int o = i / 50, t = i - o * 50;
    float acc = bp[o];
    #pragma unroll
    for (int ci = 0; ci < 8; ++ci) acc += wps[o * 8 + ci] * c3[ci * 50 + t];
    const float e = eluf(acc);
    tmp4[((size_t)b * 8 + o) * 50 + t] = e;
    atomicAdd(&sacc[o], e);
    atomicAdd(&sacc[8 + o], e * e);
  }
  __syncthreads();
  if (tid < 16) atomicAdd(&stats3[tid], sacc[tid]);
}

// ===== K7: BN3 apply + maxpool5 -> seq[branch][t][batch] =====
__global__ void __launch_bounds__(256)
k_pool5(const float* __restrict__ tmp4, const float* __restrict__ sc3,
        const float* __restrict__ sh3, float* __restrict__ seq) {
  const int idx = blockIdx.x * 256 + threadIdx.x;
  if (idx >= NB * 80) return;
  const int b = idx & (NB - 1);
  const int r = idx >> 11;
  const int c = r / 10, t = r - c * 10;
  const float* p = tmp4 + ((size_t)b * 8 + c) * 50 + t * 5;
  const float s = sc3[c], sh = sh3[c];
  float m = p[0] * s + sh;
  for (int q = 1; q < 5; ++q) { const float v = p[q] * s + sh; m = v > m ? v : m; }
  seq[(c * 10 + t) * NB + b] = m;
}

// ===== K8: forward LSTM with resident state + WMMA; backward = 1 elementwise step =====
__global__ void __launch_bounds__(128)
k_lstm(const float* __restrict__ seq, const float* __restrict__ wih,
       const float* __restrict__ whh, const float* __restrict__ bih,
       const float* __restrict__ bhh, float* __restrict__ feats) {
  extern __shared__ char smraw[];
  _Float16* Bm = (_Float16*)smraw;         // 480*BSTR  (Whh^T, N-major, K padded to 128)
  _Float16* hM = Bm + 480 * BSTR;          // 32*HSTR
  float* cM = (float*)(hM + 32 * HSTR);    // 32*120
  float* zM = cM + 32 * 120;               // 32*480
  float* wW = zM + 32 * 480;               // 480
  float* wB = wW + 480;                    // 480

  const int tid = threadIdx.x;
  const int k = blockIdx.y;
  const int batch0 = blockIdx.x * 32;
  const int base0 = (k * 2 + 0) * 480;

  for (int i = tid; i < 480; i += 128) {
    wW[i] = wih[base0 + i];
    wB[i] = bih[base0 + i] + bhh[base0 + i];
  }
  for (int i = tid; i < 480 * BSTR; i += 128) Bm[i] = (_Float16)0.f;
  for (int i = tid; i < 32 * HSTR; i += 128) hM[i] = (_Float16)0.f;
  for (int i = tid; i < 32 * 120; i += 128) cM[i] = 0.f;
  __syncthreads();
  const float* whh0 = whh + (size_t)base0 * 120;
  for (int i = tid; i < 480 * 120; i += 128) {
    const int n = i / 120, kk = i - n * 120;
    Bm[n * BSTR + kk] = (_Float16)whh0[i];
  }
  __syncthreads();

  const int lane = tid & 31;
  const int w = tid >> 5;          // 4 waves
  const int lr = lane & 15;
  const int hi = lane >> 4;
  const int mi = w >> 1;           // M-tile (0..1)
  const int nb = (w & 1) * 15;     // first of 15 N-tiles

  const v8f vzero = {0.f, 0.f, 0.f, 0.f, 0.f, 0.f, 0.f, 0.f};

  for (int t = 0; t < 10; ++t) {
    // ---- z = h @ Whh^T via v_wmma_f32_16x16x32_f16 ----
    v8f acc[15];
    #pragma unroll
    for (int q = 0; q < 15; ++q) acc[q] = vzero;
    #pragma unroll
    for (int kb = 0; kb < 4; ++kb) {
      // A fragment: row = mi*16+lr; halves 0..7 = K kb*32+hi*8.., 8..15 = +16
      const _Float16* hrow = hM + (mi * 16 + lr) * HSTR + kb * 32 + hi * 8;
      const v8h a0 = *(const v8h*)(hrow);
      const v8h a1 = *(const v8h*)(hrow + 16);
      v16h a;
      #pragma unroll
      for (int e = 0; e < 8; ++e) { a[e] = a0[e]; a[8 + e] = a1[e]; }
      #pragma unroll
      for (int q = 0; q < 15; ++q) {
        // B fragment: col = (nb+q)*16+lr; halves 0..7 = K kb*32+hi*8.., 8..15 = +16
        const _Float16* brow = Bm + ((nb + q) * 16 + lr) * BSTR + kb * 32 + hi * 8;
        const v8h b0 = *(const v8h*)(brow);
        const v8h b1 = *(const v8h*)(brow + 16);
        v16h bf;
        #pragma unroll
        for (int e = 0; e < 8; ++e) { bf[e] = b0[e]; bf[8 + e] = b1[e]; }
        acc[q] = __builtin_amdgcn_wmma_f32_16x16x32_f16(false, a, false, bf,
                                                        (short)0, acc[q], false, false);
      }
    }
    #pragma unroll
    for (int q = 0; q < 15; ++q) {
      float* zrow = zM + (mi * 16 + hi * 8) * 480 + (nb + q) * 16 + lr;
      #pragma unroll
      for (int r = 0; r < 8; ++r) zrow[r * 480] = acc[q][r];
    }
    __syncthreads();
    // ---- gates (i,f,g,o), update c,h ----
    for (int cell = tid; cell < 32 * 120; cell += 128) {
      const int row = cell / 120, j = cell - row * 120;
      const float xt = seq[(k * 10 + t) * NB + batch0 + row];
      const float* zr = zM + row * 480;
      const float zi = zr[j]       + wW[j]       * xt + wB[j];
      const float zf = zr[120 + j] + wW[120 + j] * xt + wB[120 + j];
      const float zg = zr[240 + j] + wW[240 + j] * xt + wB[240 + j];
      const float zo = zr[360 + j] + wW[360 + j] * xt + wB[360 + j];
      float c = cM[cell];
      c = sigf(zf) * c + sigf(zi) * tanhf(zg);
      const float h = sigf(zo) * tanhf(c);
      cM[cell] = c;
      hM[row * HSTR + j] = (_Float16)h;
    }
    __syncthreads();
  }

  // hf out + backward direction (single step from zero state: f-gate irrelevant)
  const int base1 = (k * 2 + 1) * 480;
  for (int cell = tid; cell < 32 * 120; cell += 128) {
    const int row = cell / 120, j = cell - row * 120;
    const int b = batch0 + row;
    feats[((size_t)(k * 240) + j) * NB + b] = (float)hM[row * HSTR + j];
    const float x9 = seq[(k * 10 + 9) * NB + b];
    const float zi = wih[base1 + j]       * x9 + bih[base1 + j]       + bhh[base1 + j];
    const float zg = wih[base1 + 240 + j] * x9 + bih[base1 + 240 + j] + bhh[base1 + 240 + j];
    const float zo = wih[base1 + 360 + j] * x9 + bih[base1 + 360 + j] + bhh[base1 + 360 + j];
    const float cb = sigf(zi) * tanhf(zg);
    feats[((size_t)(k * 240) + 120 + j) * NB + b] = sigf(zo) * tanhf(cb);
  }
}

// ===== K9: per-(branch,feature) BN stats over batch, fold into head weights =====
__global__ void __launch_bounds__(256)
k_featstats(const float* __restrict__ feats, const float* __restrict__ g,
            const float* __restrict__ beta, const float* __restrict__ hw,
            float* __restrict__ A, float* __restrict__ Cacc) {
  __shared__ float r1[256], r2[256];
  const int kd = blockIdx.x;
  const int k = kd / 240;
  const float* col = feats + (size_t)kd * NB;
  float s = 0.f, ss = 0.f;
  for (int i = threadIdx.x; i < NB; i += 256) { const float v = col[i]; s += v; ss += v * v; }
  r1[threadIdx.x] = s; r2[threadIdx.x] = ss;
  __syncthreads();
  for (int st = 128; st > 0; st >>= 1) {
    if (threadIdx.x < st) {
      r1[threadIdx.x] += r1[threadIdx.x + st];
      r2[threadIdx.x] += r2[threadIdx.x + st];
    }
    __syncthreads();
  }
  if (threadIdx.x == 0) {
    const float m = r1[0] / (float)NB;
    const float v = r2[0] / (float)NB - m * m;
    const float sc = g[kd] * rsqrtf(v);
    A[kd] = sc * hw[kd];
    atomicAdd(&Cacc[k], (beta[kd] - m * sc) * hw[kd]);
  }
}

// ===== K10: head dot + relu, pre-output stats =====
__global__ void __launch_bounds__(256)
k_head(const float* __restrict__ feats, const float* __restrict__ A,
       const float* __restrict__ Cacc, const float* __restrict__ hb,
       float* __restrict__ zbuf, float* __restrict__ statsO) {
  __shared__ float sacc[2];
  if (threadIdx.x < 2) sacc[threadIdx.x] = 0.f;
  __syncthreads();
  const int k = blockIdx.y;
  const int b = blockIdx.x * 256 + threadIdx.x;
  float acc = Cacc[k] + hb[k];
  const float* fk = feats + (size_t)k * 240 * NB + b;
  const float* Ak = A + k * 240;
  for (int d = 0; d < 240; ++d) acc += fk[(size_t)d * NB] * Ak[d];
  const float h = acc > 0.f ? acc : 0.f;
  zbuf[k * NB + b] = h;
  atomicAdd(&sacc[0], h);
  atomicAdd(&sacc[1], h * h);
  __syncthreads();
  if (threadIdx.x == 0) { atomicAdd(&statsO[k], sacc[0]); atomicAdd(&statsO[8 + k], sacc[1]); }
}

// ===== K12: output BN + linear(8->1) + sigmoid =====
__global__ void __launch_bounds__(256)
k_final(const float* __restrict__ zbuf, const float* __restrict__ scO,
        const float* __restrict__ shO, const float* __restrict__ ow,
        const float* __restrict__ ob, float* __restrict__ out) {
  const int b = blockIdx.x * 256 + threadIdx.x;
  if (b >= NB) return;
  float acc = ob[0];
  #pragma unroll
  for (int kk = 0; kk < 8; ++kk)
    acc += (zbuf[kk * NB + b] * scO[kk] + shO[kk]) * ow[kk];
  out[b] = 1.f / (1.f + expf(-acc));
}

extern "C" void kernel_launch(void* const* d_in, const int* in_sizes, int n_in,
                              void* d_out, int out_size, void* d_ws, size_t ws_size,
                              hipStream_t stream) {
  (void)in_sizes; (void)n_in; (void)out_size; (void)ws_size;
  const float* x    = (const float*)d_in[0];
  const float* w1   = (const float*)d_in[1];
  const float* b1   = (const float*)d_in[2];
  const float* g1   = (const float*)d_in[3];
  const float* be1  = (const float*)d_in[4];
  const float* w2   = (const float*)d_in[5];
  const float* b2   = (const float*)d_in[6];
  const float* g2   = (const float*)d_in[7];
  const float* be2  = (const float*)d_in[8];
  const float* w3   = (const float*)d_in[9];
  const float* b3   = (const float*)d_in[10];
  const float* wp   = (const float*)d_in[11];
  const float* bp   = (const float*)d_in[12];
  const float* g3   = (const float*)d_in[13];
  const float* be3  = (const float*)d_in[14];
  const float* wih  = (const float*)d_in[15];
  const float* whh  = (const float*)d_in[16];
  const float* bihp = (const float*)d_in[17];
  const float* bhhp = (const float*)d_in[18];
  const float* gl   = (const float*)d_in[19];
  const float* bl   = (const float*)d_in[20];
  const float* hw   = (const float*)d_in[21];
  const float* hb   = (const float*)d_in[22];
  const float* go   = (const float*)d_in[23];
  const float* bo   = (const float*)d_in[24];
  const float* ow   = (const float*)d_in[25];
  const float* ob   = (const float*)d_in[26];
  float* ws  = (float*)d_ws;
  float* out = (float*)d_out;

  const size_t SM1 = (12800 + 404 + 8) * sizeof(float);
  const size_t SM3 = (12800 + 404 + 512 + 1600 + 16) * sizeof(float);
  const size_t SM5 = (1600 + 400 + 400 + 200 + 64 + 16) * sizeof(float);
  const size_t SML = (size_t)480 * BSTR * 2 + (size_t)32 * HSTR * 2 +
                     (size_t)32 * 120 * 4 + (size_t)32 * 480 * 4 + 960 * 4;

  hipFuncSetAttribute((const void*)k_conv1_stats, hipFuncAttributeMaxDynamicSharedMemorySize, (int)SM1);
  hipFuncSetAttribute((const void*)k_conv12,      hipFuncAttributeMaxDynamicSharedMemorySize, (int)SM3);
  hipFuncSetAttribute((const void*)k_stage2,      hipFuncAttributeMaxDynamicSharedMemorySize, (int)SM5);
  hipFuncSetAttribute((const void*)k_lstm,        hipFuncAttributeMaxDynamicSharedMemorySize, (int)SML);

  // zero all global stats accumulators (first 128 floats of ws)
  hipMemsetAsync(d_ws, 0, 128 * sizeof(float), stream);

  k_conv1_stats<<<NB, 256, SM1, stream>>>(x, w1, b1, ws + OFF_S1);
  k_finalize<<<1, 32, 0, stream>>>(ws + OFF_S1, g1, be1, ws + OFF_SC1, ws + OFF_SH1, 4, 26214400.f);
  k_conv12<<<NB, 256, SM3, stream>>>(x, w1, b1, ws + OFF_SC1, ws + OFF_SH1, w2, b2,
                                     ws + OFF_TMP2, ws + OFF_S2);
  k_finalize<<<1, 32, 0, stream>>>(ws + OFF_S2, g2, be2, ws + OFF_SC2, ws + OFF_SH2, 8, 409600.f);
  k_stage2<<<NB, 256, SM5, stream>>>(ws + OFF_TMP2, ws + OFF_SC2, ws + OFF_SH2,
                                     w3, b3, wp, bp, ws + OFF_TMP4, ws + OFF_S3);
  k_finalize<<<1, 32, 0, stream>>>(ws + OFF_S3, g3, be3, ws + OFF_SC3, ws + OFF_SH3, 8, 102400.f);
  k_pool5<<<(NB * 80 + 255) / 256, 256, 0, stream>>>(ws + OFF_TMP4, ws + OFF_SC3, ws + OFF_SH3,
                                                     ws + OFF_SEQ);
  k_lstm<<<dim3(NB / 32, 8), 128, SML, stream>>>(ws + OFF_SEQ, wih, whh, bihp, bhhp, ws + OFF_FEATS);
  k_featstats<<<8 * 240, 256, 0, stream>>>(ws + OFF_FEATS, gl, bl, hw, ws + OFF_A, ws + OFF_CACC);
  k_head<<<dim3(NB / 256, 8), 256, 0, stream>>>(ws + OFF_FEATS, ws + OFF_A, ws + OFF_CACC, hb,
                                                ws + OFF_ZB, ws + OFF_SO);
  k_finalize<<<1, 32, 0, stream>>>(ws + OFF_SO, go, bo, ws + OFF_SCO, ws + OFF_SHO, 8, 2048.f);
  k_final<<<NB / 256, 256, 0, stream>>>(ws + OFF_ZB, ws + OFF_SCO, ws + OFF_SHO, ow, ob, out);
}
